// MultiHeadAttentionV2_18494129177040
// MI455X (gfx1250) — matrix operand order repaired and checked by reading, not compiled, not run on hardware
//
#include <hip/hip_runtime.h>
#include <stdint.h>

typedef __attribute__((ext_vector_type(16))) __bf16 v16bf;
typedef __attribute__((ext_vector_type(8)))  float  v8f;

#define WMMA_BF16(A_, B_, C_) \
  __builtin_amdgcn_wmma_f32_16x16x32_bf16(false, (A_), false, (B_), (short)0, (C_), false, false)

// Async global->LDS DMA copy, 16 bytes per lane (CDNA5, tracked by ASYNCcnt).
// GVS addressing: 64-bit uniform base in SGPR pair + per-lane 32-bit byte offset.
__device__ __forceinline__ void async_b128(uint32_t lds_byte_addr,
                                           const void* base, uint32_t byte_off) {
  asm volatile("global_load_async_to_lds_b128 %0, %1, %2 offset:0"
               :: "v"(lds_byte_addr), "v"(byte_off),
                  "s"((uint64_t)(uintptr_t)base)
               : "memory");
}
__device__ __forceinline__ void wait_async0() {
  asm volatile("s_wait_asynccnt 0x0" ::: "memory");
}

__device__ __forceinline__ uint16_t f2bf(float f) {
  union { float f; uint32_t u; } c; c.f = f;
  uint32_t u = c.u;
  u += 0x7FFFu + ((u >> 16) & 1u);   // round-to-nearest-even
  return (uint16_t)(u >> 16);
}

union FragU { uint4 q[2]; v16bf v; };

__device__ __forceinline__ v16bf load_frag(const uint16_t* p0, const uint16_t* p1) {
  FragU f;
  f.q[0] = *reinterpret_cast<const uint4*>(p0);
  f.q[1] = *reinterpret_cast<const uint4*>(p1);
  return f.v;
}

__device__ __forceinline__ float red_max16(float v) {
#pragma unroll
  for (int m = 1; m < 16; m <<= 1) v = fmaxf(v, __shfl_xor(v, m, 16));
  return v;
}
__device__ __forceinline__ float red_sum16(float v) {
#pragma unroll
  for (int m = 1; m < 16; m <<= 1) v += __shfl_xor(v, m, 16);
  return v;
}

// ---------------------------------------------------------------- conversion
__global__ void cvt_f32_bf16x4(const float* __restrict__ in,
                               uint16_t* __restrict__ out, int n4) {
  int i = blockIdx.x * blockDim.x + threadIdx.x;
  if (i < n4) {
    float4 f = reinterpret_cast<const float4*>(in)[i];
    uint2 u;
    u.x = (uint32_t)f2bf(f.x) | ((uint32_t)f2bf(f.y) << 16);
    u.y = (uint32_t)f2bf(f.z) | ((uint32_t)f2bf(f.w) << 16);
    reinterpret_cast<uint2*>(out)[i] = u;
  }
}

// ------------------------------------------------------- bf16 WMMA GEMM
// C[M,N] = A[M,K] * B[K,N], A/B bf16 row-major, f32 accumulate.
// WG = 256 thr = 8 waves; WG tile 128x128; wave tile 64(M) x 32(N).
// Double-buffered LDS: A staged via async DMA, B pipelined through registers
// (needs transpose for the WMMA B-fragment layout).
template <bool BF16OUT>
__global__ __launch_bounds__(256) void gemm_bf16(const uint16_t* __restrict__ A,
                                                 const uint16_t* __restrict__ B,
                                                 void* __restrict__ Cout,
                                                 int M, int N, int K) {
  __shared__ uint16_t As[2][128][40];   // [m][k], row-padded to 40 halves (80B)
  __shared__ uint16_t Bs[2][128][40];   // transposed: [n][k]

  const int tid  = threadIdx.x;
  const int lane = tid & 31;
  const int wid  = tid >> 5;
  const int wm   = (wid & 1) * 64;
  const int wn   = (wid >> 1) * 32;
  const int laneN = lane & 15, laneH = lane >> 4;
  const int rowBase = blockIdx.y * 128;
  const int colBase = blockIdx.x * 128;

  v8f acc[4][2];
#pragma unroll
  for (int mi = 0; mi < 4; ++mi)
#pragma unroll
    for (int ni = 0; ni < 2; ++ni) acc[mi][ni] = (v8f){0,0,0,0,0,0,0,0};

  const int aRow = tid >> 1, aChunk = (tid & 1) * 16;   // 128 rows x 2 chunks
  const int bK   = tid & 31, bChunk = (tid >> 5) * 16;  // 32 k x 8 n-chunks

  auto stageA = [&](int k0, int buf) {   // async DMA: 2 x 16B per thread
    uint32_t lds = (uint32_t)(uintptr_t)&As[buf][aRow][aChunk];
    uint32_t off = (uint32_t)(((size_t)(rowBase + aRow) * K + k0 + aChunk) * 2);
    async_b128(lds, A, off);
    async_b128(lds + 16, A, off + 16);
  };
  auto loadB = [&](int k0, uint4& r0, uint4& r1) {
    const uint16_t* bp = B + (size_t)(k0 + bK) * N + colBase + bChunk;
    r0 = *reinterpret_cast<const uint4*>(bp);
    r1 = *reinterpret_cast<const uint4*>(bp + 8);
  };
  auto scatterB = [&](int buf, uint4 r0, uint4 r1) {
    uint16_t tmp[16];
    *reinterpret_cast<uint4*>(tmp)     = r0;
    *reinterpret_cast<uint4*>(tmp + 8) = r1;
#pragma unroll
    for (int i = 0; i < 16; ++i) Bs[buf][bChunk + i][bK] = tmp[i];
  };

  // prologue: stage tile 0
  uint4 b0, b1;
  stageA(0, 0);
  loadB(0, b0, b1);
  scatterB(0, b0, b1);
  wait_async0();
  __syncthreads();

  int buf = 0;
  for (int k0 = 0; k0 < K; k0 += 32) {
    const int nxt = buf ^ 1;
    const bool has_next = (k0 + 32 < K);
    if (has_next) {                 // overlap next tile's HBM traffic with WMMA
      stageA(k0 + 32, nxt);
      loadB(k0 + 32, b0, b1);
    }

    v16bf af[4], bf[2];
#pragma unroll
    for (int mi = 0; mi < 4; ++mi) {
      const int r = wm + mi * 16 + laneN;
      af[mi] = load_frag(&As[buf][r][laneH * 8], &As[buf][r][16 + laneH * 8]);
    }
#pragma unroll
    for (int ni = 0; ni < 2; ++ni) {
      const int n = wn + ni * 16 + laneN;
      bf[ni] = load_frag(&Bs[buf][n][laneH * 8], &Bs[buf][n][16 + laneH * 8]);
    }
#pragma unroll
    for (int mi = 0; mi < 4; ++mi)
#pragma unroll
      for (int ni = 0; ni < 2; ++ni)
        acc[mi][ni] = WMMA_BF16(af[mi], bf[ni], acc[mi][ni]);

    if (has_next) {
      scatterB(nxt, b0, b1);
      wait_async0();
    }
    __syncthreads();
    buf = nxt;
  }

#pragma unroll
  for (int mi = 0; mi < 4; ++mi)
#pragma unroll
    for (int ni = 0; ni < 2; ++ni)
#pragma unroll
      for (int r = 0; r < 8; ++r) {
        const size_t row = rowBase + wm + mi * 16 + r + 8 * laneH;
        const size_t col = colBase + wn + ni * 16 + laneN;
        if (BF16OUT) ((uint16_t*)Cout)[row * N + col] = f2bf(acc[mi][ni][r]);
        else         ((float*)Cout)[row * N + col]    = acc[mi][ni][r];
      }
}

// ------------------------------------------------- flash attention (causal)
// grid: (T/64, B*H); block 128 = 4 waves; wave owns 16 q rows.
__global__ __launch_bounds__(128) void attn_flash(const uint16_t* __restrict__ qkv,
                                                  uint16_t* __restrict__ y,
                                                  int T, int C, int H) {
  __shared__ uint16_t Ks[32][72];      // [kv][d], 64 d, padded
  __shared__ uint16_t Vt[64][40];      // transposed: [d][kv]
  __shared__ uint16_t Pt[4][16][40];   // per-wave P tile [qm][kv]

  const int tid = threadIdx.x, lane = tid & 31, wid = tid >> 5;
  const int laneN = lane & 15, laneH = lane >> 4;
  const int qBlock = blockIdx.x * 64;
  const int b = blockIdx.y / H, h = blockIdx.y % H;
  const int threeC = 3 * C, hd = 64;
  const size_t base = (size_t)b * T * threeC;
  const int qRow0 = qBlock + wid * 16;

  // Q fragments (held in registers for whole kernel)
  const uint16_t* qp = qkv + base + (size_t)(qRow0 + laneN) * threeC + h * hd;
  const v16bf qa0 = load_frag(qp + laneH * 8,      qp + 16 + laneH * 8);
  const v16bf qa1 = load_frag(qp + 32 + laneH * 8, qp + 48 + laneH * 8);

  v8f o[4];
#pragma unroll
  for (int dt = 0; dt < 4; ++dt) o[dt] = (v8f){0,0,0,0,0,0,0,0};
  float mrow[8], lrow[8];
#pragma unroll
  for (int r = 0; r < 8; ++r) { mrow[r] = -3.0e38f; lrow[r] = 0.0f; }

  const float scale = 0.125f;                 // 1/sqrt(64)
  const int kvEnd = qBlock + 64;              // causal upper bound for this WG
  const int kRow = tid >> 2, kChunk = (tid & 3) * 16;   // K staging: 32x4
  const int vKv  = tid & 31, vChunk = (tid >> 5) * 16;  // V staging: 32x4

  for (int kv0 = 0; kv0 < kvEnd; kv0 += 32) {
    __syncthreads();   // previous tile fully consumed
    // stage K tile via async DMA (consumed row-major -> direct copy)
    {
      uint32_t lds = (uint32_t)(uintptr_t)&Ks[kRow][kChunk];
      uint32_t off = (uint32_t)((base + (size_t)(kv0 + kRow) * threeC + C + h * hd + kChunk) * 2);
      async_b128(lds, qkv, off);
      async_b128(lds + 16, qkv, off + 16);
    }
    // stage V tile transposed through registers
    const uint16_t* vp = qkv + base + (size_t)(kv0 + vKv) * threeC + 2 * C + h * hd + vChunk;
    uint16_t vt[16];
    *reinterpret_cast<uint4*>(vt)     = *reinterpret_cast<const uint4*>(vp);
    *reinterpret_cast<uint4*>(vt + 8) = *reinterpret_cast<const uint4*>(vp + 8);
#pragma unroll
    for (int i = 0; i < 16; ++i) Vt[vChunk + i][vKv] = vt[i];
    wait_async0();
    __syncthreads();

    // S = Q K^T for two 16-wide kv sub-tiles (contraction hd=64 -> 2 WMMAs each)
    v8f s[2];
#pragma unroll
    for (int nt = 0; nt < 2; ++nt) {
      const int kr = nt * 16 + laneN;
      v16bf kb0 = load_frag(&Ks[kr][laneH * 8],      &Ks[kr][16 + laneH * 8]);
      v16bf kb1 = load_frag(&Ks[kr][32 + laneH * 8], &Ks[kr][48 + laneH * 8]);
      v8f sa = (v8f){0,0,0,0,0,0,0,0};
      sa = WMMA_BF16(qa0, kb0, sa);
      sa = WMMA_BF16(qa1, kb1, sa);
      s[nt] = sa;
    }

    // online softmax (rows live in C/D layout: row = r + 8*laneH, col = laneN)
    float pv[2][8], alpha[8];
#pragma unroll
    for (int r = 0; r < 8; ++r) {
      const int tq = qRow0 + r + 8 * laneH;
      float s0 = s[0][r] * scale; if (kv0 + laneN > tq)      s0 = -3.0e38f;
      float s1 = s[1][r] * scale; if (kv0 + 16 + laneN > tq) s1 = -3.0e38f;
      float mt = red_max16(fmaxf(s0, s1));
      float mn = fmaxf(mrow[r], mt);
      float a  = __expf(mrow[r] - mn);
      float p0 = __expf(s0 - mn);
      float p1 = __expf(s1 - mn);
      lrow[r] = lrow[r] * a + red_sum16(p0 + p1);
      mrow[r] = mn;
      alpha[r] = a;
      pv[0][r] = p0; pv[1][r] = p1;
    }
#pragma unroll
    for (int dt = 0; dt < 4; ++dt)
#pragma unroll
      for (int r = 0; r < 8; ++r) o[dt][r] *= alpha[r];

    // re-layout P: C/D layout -> A fragment via per-wave LDS tile
#pragma unroll
    for (int nt = 0; nt < 2; ++nt)
#pragma unroll
      for (int r = 0; r < 8; ++r)
        Pt[wid][r + 8 * laneH][nt * 16 + laneN] = f2bf(pv[nt][r]);
    v16bf pa = load_frag(&Pt[wid][laneN][laneH * 8], &Pt[wid][laneN][16 + laneH * 8]);

    // O += P @ V  (4 d-tiles of 16)
#pragma unroll
    for (int dt = 0; dt < 4; ++dt) {
      const int dr = dt * 16 + laneN;
      v16bf vb = load_frag(&Vt[dr][laneH * 8], &Vt[dr][16 + laneH * 8]);
      o[dt] = WMMA_BF16(pa, vb, o[dt]);
    }
  }

  // epilogue: normalize and store y (bf16, [B*T, C] with head interleave)
#pragma unroll
  for (int r = 0; r < 8; ++r) {
    const float inv = 1.0f / lrow[r];
    const size_t row = (size_t)b * T + qRow0 + r + 8 * laneH;
#pragma unroll
    for (int dt = 0; dt < 4; ++dt)
      y[row * C + h * hd + dt * 16 + laneN] = f2bf(o[dt][r] * inv);
  }
}

// ---------------------------------------------------------------- launcher
extern "C" void kernel_launch(void* const* d_in, const int* in_sizes, int n_in,
                              void* d_out, int out_size, void* d_ws, size_t ws_size,
                              hipStream_t stream) {
  const float* x      = (const float*)d_in[0];
  const float* w_attn = (const float*)d_in[1];
  const float* w_proj = (const float*)d_in[2];
  float* out = (float*)d_out;

  const int Bb = 2, T = 2048, C = 1024, H = 16;
  const int M = Bb * T;            // 4096
  const int threeC = 3 * C;        // 3072

  // workspace carve-up (bf16 buffers), 256B aligned
  auto align256 = [](size_t v) { return (v + 255) & ~(size_t)255; };
  uint8_t* ws = (uint8_t*)d_ws;
  size_t off = 0;
  uint16_t* xb   = (uint16_t*)(ws + off); off = align256(off + (size_t)M * C * 2);
  uint16_t* wab  = (uint16_t*)(ws + off); off = align256(off + (size_t)C * threeC * 2);
  uint16_t* wpb  = (uint16_t*)(ws + off); off = align256(off + (size_t)C * C * 2);
  uint16_t* qkvb = (uint16_t*)(ws + off); off = align256(off + (size_t)M * threeC * 2);
  uint16_t* yb   = (uint16_t*)(ws + off);

  // fp32 -> bf16 conversions
  {
    int n4 = M * C / 4;
    cvt_f32_bf16x4<<<(n4 + 255) / 256, 256, 0, stream>>>(x, xb, n4);
    n4 = C * threeC / 4;
    cvt_f32_bf16x4<<<(n4 + 255) / 256, 256, 0, stream>>>(w_attn, wab, n4);
    n4 = C * C / 4;
    cvt_f32_bf16x4<<<(n4 + 255) / 256, 256, 0, stream>>>(w_proj, wpb, n4);
  }

  // qkv = x @ w_attn  (bf16 out)
  gemm_bf16<true><<<dim3(threeC / 128, M / 128), 256, 0, stream>>>(xb, wab, qkvb, M, threeC, C);

  // causal flash attention -> y (bf16)
  attn_flash<<<dim3(T / 64, Bb * H), 128, 0, stream>>>(qkvb, yb, T, C, H);

  // out = y @ w_proj  (f32 out)
  gemm_bf16<false><<<dim3(C / 128, M / 128), 256, 0, stream>>>(yb, wpb, out, M, C, C);
}
